// DeTrAttention_25340307047006
// MI455X (gfx1250) — compile-verified, matrix operand order checked by reading, not run on hardware
//
#include <hip/hip_runtime.h>
#include <hip/hip_bf16.h>

typedef __bf16 v16bf __attribute__((ext_vector_type(16)));
typedef __bf16 v8bf  __attribute__((ext_vector_type(8)));
typedef __bf16 v4bf  __attribute__((ext_vector_type(4)));
typedef float  v8f   __attribute__((ext_vector_type(8)));

#define DEV __device__ __forceinline__

// f32 -> bf16, round-to-nearest-even
DEV __bf16 f2b(float f) {
    union { float f; unsigned int u; } c; c.f = f;
    unsigned int u = c.u + 0x7fffu + ((c.u >> 16) & 1u);
    unsigned short h = (unsigned short)(u >> 16);
    __bf16 r;
    __builtin_memcpy(&r, &h, sizeof(r));
    return r;
}

// Build a 16x16x32 bf16 WMMA fragment for this lane from two contiguous
// 8-element (16B) runs: K = kb..kb+7 (VGPRs 0-3) and kb+16..kb+23 (VGPRs 4-7).
DEV v16bf make_frag(const __bf16* lo, const __bf16* hi) {
    v8bf a = *(const v8bf*)lo;
    v8bf b = *(const v8bf*)hi;
    v16bf r;
#pragma unroll
    for (int i = 0; i < 8; ++i) { r[i] = a[i]; r[i + 8] = b[i]; }
    return r;
}

DEV v8f wmma_bf16(v16bf a, v16bf b, v8f c) {
    return __builtin_amdgcn_wmma_f32_16x16x32_bf16(
        /*neg_a=*/false, a, /*neg_b=*/false, b,
        /*c_mod=*/(short)0, c, /*reuse_a=*/false, /*reuse_b=*/false);
}

// CDNA5 async global->LDS copy (16B per lane), tracked by ASYNCcnt.
DEV void async_copy_b128(unsigned lds_byte_off, unsigned global_byte_off,
                         unsigned long long base) {
    asm volatile("global_load_async_to_lds_b128 %0, %1, %2"
                 :: "v"(lds_byte_off), "v"(global_byte_off), "s"(base)
                 : "memory");
}
DEV void wait_async0() {
    asm volatile("s_wait_asynccnt 0x0" ::: "memory");
}
DEV unsigned lds_off(const void* p) {
    return (unsigned)(unsigned long long)p;  // low 32 bits of LDS aperture addr
}

// ---------------------------------------------------------------------------
// GEMM: out[M,N] = A[M,K] @ W[K,N] + bias[N]
//   A_F32:  A is f32 (register-pipelined, converted to bf16 into LDS);
//           else A is bf16 (async global->LDS copy).
//   OUT_F32: store f32; else store bf16.
// Block tile 128x128, K-step 32. 8 waves; wave = (wm in 0..1) x (wn in 0..3),
// each wave computes a 64x32 subtile = 4x2 WMMA tiles.
// K loop is software-pipelined: tile t+1 fetched to registers while tile t
// is consumed by WMMAs.
// ---------------------------------------------------------------------------
template <bool A_F32, bool OUT_F32>
__global__ __launch_bounds__(256)
void gemm_bias_kernel(const void* __restrict__ Ain,
                      const float* __restrict__ W,
                      const float* __restrict__ bias,
                      void* __restrict__ outp,
                      int M, int N, int K)
{
    __shared__ __attribute__((aligned(16))) __bf16 sA[128][40]; // [m][k], 80B pitch
    __shared__ __attribute__((aligned(16))) __bf16 sB[128][40]; // [n][k] (W transposed)

    const int tid  = threadIdx.x;
    const int lane = tid & 31;
    const int wave = tid >> 5;
    const int wm   = wave >> 2;           // 0..1 : 64 rows
    const int wn   = wave & 3;            // 0..3 : 32 cols
    const int lm   = lane & 15;
    const int kb   = (lane & 16) ? 8 : 0;
    const int hi8  = (lane >> 4) << 3;

    const int m0 = blockIdx.y * 128;
    const int n0 = blockIdx.x * 128;

    v8f acc[4][2];
#pragma unroll
    for (int tm = 0; tm < 4; ++tm)
#pragma unroll
        for (int tn = 0; tn < 2; ++tn)
#pragma unroll
            for (int j = 0; j < 8; ++j) acc[tm][tn][j] = 0.f;

    // register staging for the pipelined fetch
    float4 ra[4];   // A chunks (A_F32 path)
    float4 rw[4];   // W chunks

    auto fetchW = [&](int k0) {
#pragma unroll
        for (int u = 0; u < 4; ++u) {
            int c = tid + (u << 8);
            int kr = c >> 5, n4 = (c & 31) << 2;
            rw[u] = *(const float4*)(W + (size_t)(k0 + kr) * N + n0 + n4);
        }
    };
    auto fetchA = [&](int k0) {
        if constexpr (A_F32) {
            const float* A = (const float*)Ain;
#pragma unroll
            for (int u = 0; u < 4; ++u) {
                int c = tid + (u << 8);
                int row = c >> 3, c4 = (c & 7) << 2;
                ra[u] = *(const float4*)(A + (size_t)(m0 + row) * K + k0 + c4);
            }
        }
    };

    fetchW(0);
    fetchA(0);

    const unsigned ldsA0 = lds_off(&sA[0][0]);
    const unsigned long long abase = (unsigned long long)Ain;

    for (int k0 = 0; k0 < K; k0 += 32) {
        __syncthreads();   // previous compute done reading LDS

        // --- stage A tile (128 x 32) ---
        if constexpr (A_F32) {
#pragma unroll
            for (int u = 0; u < 4; ++u) {
                int c = tid + (u << 8);
                int row = c >> 3, c4 = (c & 7) << 2;
                v4bf p;
                p[0] = f2b(ra[u].x); p[1] = f2b(ra[u].y);
                p[2] = f2b(ra[u].z); p[3] = f2b(ra[u].w);
                *(v4bf*)&sA[row][c4] = p;
            }
        } else {
            // async global->LDS copy of bf16 A tile (16B per lane per op)
#pragma unroll
            for (int u = 0; u < 2; ++u) {
                int c = tid + (u << 8);
                int row = c >> 2, c8 = (c & 3) << 3;
                unsigned dst = ldsA0 + (unsigned)(row * 40 + c8) * 2u;
                unsigned src = (unsigned)(((size_t)(m0 + row) * K + k0 + c8) * 2u);
                async_copy_b128(dst, src, abase);
            }
        }
        // --- stage W tile transposed (32 x 128 -> sB[n][k]) ---
#pragma unroll
        for (int u = 0; u < 4; ++u) {
            int c = tid + (u << 8);
            int kr = c >> 5, n4 = (c & 31) << 2;
            sB[n4 + 0][kr] = f2b(rw[u].x);
            sB[n4 + 1][kr] = f2b(rw[u].y);
            sB[n4 + 2][kr] = f2b(rw[u].z);
            sB[n4 + 3][kr] = f2b(rw[u].w);
        }
        if constexpr (!A_F32) wait_async0();
        __syncthreads();   // staged tile visible to all waves

        // --- prefetch next tile into registers (overlaps the WMMAs below) ---
        if (k0 + 32 < K) {
            fetchW(k0 + 32);
            fetchA(k0 + 32);
        }

        // --- compute ---
        v16bf af[4], bfv[2];
#pragma unroll
        for (int tm = 0; tm < 4; ++tm) {
            const __bf16* p = &sA[(wm << 6) + (tm << 4) + lm][0];
            af[tm] = make_frag(p + kb, p + kb + 16);
        }
#pragma unroll
        for (int tn = 0; tn < 2; ++tn) {
            const __bf16* p = &sB[(wn << 5) + (tn << 4) + lm][0];
            bfv[tn] = make_frag(p + kb, p + kb + 16);
        }
#pragma unroll
        for (int tm = 0; tm < 4; ++tm)
#pragma unroll
            for (int tn = 0; tn < 2; ++tn)
                acc[tm][tn] = wmma_bf16(af[tm], bfv[tn], acc[tm][tn]);
    }

    // --- epilogue: bias + store (C layout: lane -> col lm, vgpr j -> row j+hi8)
#pragma unroll
    for (int tn = 0; tn < 2; ++tn) {
        const int col = n0 + (wn << 5) + (tn << 4) + lm;
        const float bs = bias[col];
#pragma unroll
        for (int tm = 0; tm < 4; ++tm) {
#pragma unroll
            for (int j = 0; j < 8; ++j) {
                const int row = m0 + (wm << 6) + (tm << 4) + j + hi8;
                const float v = acc[tm][tn][j] + bs;
                if constexpr (OUT_F32)
                    ((float*)outp)[(size_t)row * N + col] = v;
                else
                    ((__bf16*)outp)[(size_t)row * N + col] = f2b(v);
            }
        }
    }
}

// ---------------------------------------------------------------------------
// Flash attention: one block per (batch*head, 64 query rows).
// 8 waves: (wm in 0..3) x (wn in 0..1). Wave owns 16 q-rows x 32 out-cols.
// K tile staged to LDS via async copy (kills the 4x redundant global reads);
// V tile transposed into LDS; online softmax state in threads 0..63.
// ---------------------------------------------------------------------------
__global__ __launch_bounds__(256)
void attention_kernel(const __bf16* __restrict__ qp,
                      const __bf16* __restrict__ kp,
                      const __bf16* __restrict__ vp,
                      __bf16* __restrict__ ctx)
{
    constexpr int S = 2048, H = 1024, HD = 64;
    constexpr float scale = 0.125f;  // 1/sqrt(64)

    __shared__ __attribute__((aligned(16))) float  sS[64][65];   // raw scores
    __shared__ __attribute__((aligned(16))) __bf16 sP[64][72];   // probs (A of P@V)
    __shared__ __attribute__((aligned(16))) __bf16 sVT[64][72];  // V^T: [hd][key]
    __shared__ __attribute__((aligned(16))) __bf16 sK[64][72];   // K tile [key][hd]
    __shared__ float sRow[64];                                   // corr / 1/l

    const int tid  = threadIdx.x;
    const int lane = tid & 31;
    const int wave = tid >> 5;
    const int wm   = wave >> 1;           // 0..3 : 16 q rows
    const int wn   = wave & 1;            // 0..1 : 32 cols
    const int lm   = lane & 15;
    const int kb   = (lane & 16) ? 8 : 0;
    const int hi8  = (lane >> 4) << 3;

    const int q0 = blockIdx.x << 6;
    const int bh = blockIdx.y;
    const size_t hoff = (size_t)(bh >> 4) * S * H + (size_t)(bh & 15) * HD;
    const __bf16* Q  = qp + hoff;
    const __bf16* Kb = kp + hoff;
    const __bf16* Vb = vp + hoff;
    __bf16*       Ob = ctx + hoff;

    const unsigned long long kbase = (unsigned long long)Kb;
    const unsigned ldsK0 = lds_off(&sK[0][0]);

    // Q fragments for this wave's 16 rows (persist across the whole key loop)
    v16bf qf[2];
    {
        const __bf16* qr = Q + (size_t)(q0 + (wm << 4) + lm) * H;
        qf[0] = make_frag(qr + kb,      qr + kb + 16);
        qf[1] = make_frag(qr + 32 + kb, qr + 32 + kb + 16);
    }

    v8f o[2];
#pragma unroll
    for (int tn = 0; tn < 2; ++tn)
#pragma unroll
        for (int j = 0; j < 8; ++j) o[tn][j] = 0.f;

    float mrow = -1e30f, lrow = 0.f;

    for (int s0 = 0; s0 < S; s0 += 64) {
        // --- async stage K tile: sK[key][hd] (raw bf16 rows) ---
#pragma unroll
        for (int u = 0; u < 2; ++u) {
            int c = tid + (u << 8);
            int key = c >> 3, hd8 = (c & 7) << 3;
            unsigned dst = ldsK0 + (unsigned)(key * 72 + hd8) * 2u;
            unsigned src = (unsigned)(((size_t)(s0 + key) * H + hd8) * 2u);
            async_copy_b128(dst, src, kbase);
        }
        // --- stage V tile transposed: sVT[hd][key] (loads hoisted) ---
        v8bf vv[2];
#pragma unroll
        for (int u = 0; u < 2; ++u) {
            int c = tid + (u << 8);
            int key = c >> 3, hd8 = (c & 7) << 3;
            vv[u] = *(const v8bf*)(Vb + (size_t)(s0 + key) * H + hd8);
        }
        if (s0 + 64 < S) {  // prefetch next K/V rows into L2/WGP$
            __builtin_prefetch(Vb + (size_t)(s0 + 64 + (tid & 63)) * H, 0, 0);
            __builtin_prefetch(Kb + (size_t)(s0 + 64 + (tid & 63)) * H, 0, 0);
        }
#pragma unroll
        for (int u = 0; u < 2; ++u) {
            int c = tid + (u << 8);
            int key = c >> 3, hd8 = (c & 7) << 3;
#pragma unroll
            for (int j = 0; j < 8; ++j) sVT[hd8 + j][key] = vv[u][j];
        }
        wait_async0();
        __syncthreads();   // sK / sVT ready

        // --- scores: S = Q K^T * scale (K rows in LDS are the B^T layout) ---
#pragma unroll
        for (int tn = 0; tn < 2; ++tn) {
            const __bf16* kr = &sK[(wn << 5) + (tn << 4) + lm][0];
            v16bf b0 = make_frag(kr + kb,      kr + kb + 16);
            v16bf b1 = make_frag(kr + 32 + kb, kr + 32 + kb + 16);
            v8f sc;
#pragma unroll
            for (int j = 0; j < 8; ++j) sc[j] = 0.f;
            sc = wmma_bf16(qf[0], b0, sc);
            sc = wmma_bf16(qf[1], b1, sc);
#pragma unroll
            for (int j = 0; j < 8; ++j)
                sS[(wm << 4) + j + hi8][(wn << 5) + (tn << 4) + lm] =
                    sc[j] * scale;
        }
        __syncthreads();

        // --- online softmax: thread t owns query row t ---
        if (tid < 64) {
            float mx = -1e30f;
            for (int j = 0; j < 64; ++j) mx = fmaxf(mx, sS[tid][j]);
            float mnew = fmaxf(mrow, mx);
            float corr = __expf(mrow - mnew);
            float sum = 0.f;
            for (int j = 0; j < 64; ++j) {
                float p = __expf(sS[tid][j] - mnew);
                sum += p;
                sP[tid][j] = f2b(p);
            }
            lrow = lrow * corr + sum;
            sRow[tid] = corr;
            mrow = mnew;
        }
        __syncthreads();

        // --- rescale O and accumulate P @ V ---
        float cr[8];
#pragma unroll
        for (int j = 0; j < 8; ++j) cr[j] = sRow[(wm << 4) + j + hi8];

        v16bf pa0, pa1;
        {
            const __bf16* pr = &sP[(wm << 4) + lm][0];
            pa0 = make_frag(pr + kb,      pr + kb + 16);
            pa1 = make_frag(pr + 32 + kb, pr + 32 + kb + 16);
        }
#pragma unroll
        for (int tn = 0; tn < 2; ++tn) {
            const __bf16* vr = &sVT[(wn << 5) + (tn << 4) + lm][0];
            v16bf vb0 = make_frag(vr + kb,      vr + kb + 16);
            v16bf vb1 = make_frag(vr + 32 + kb, vr + 32 + kb + 16);
#pragma unroll
            for (int j = 0; j < 8; ++j) o[tn][j] *= cr[j];
            o[tn] = wmma_bf16(pa0, vb0, o[tn]);
            o[tn] = wmma_bf16(pa1, vb1, o[tn]);
        }
        __syncthreads();
    }

    // --- final 1/l normalization, store merged-head context (bf16) ---
    if (tid < 64) sRow[tid] = 1.0f / lrow;
    __syncthreads();
    float inv[8];
#pragma unroll
    for (int j = 0; j < 8; ++j) inv[j] = sRow[(wm << 4) + j + hi8];
#pragma unroll
    for (int tn = 0; tn < 2; ++tn)
#pragma unroll
        for (int j = 0; j < 8; ++j) {
            const int row = q0 + (wm << 4) + j + hi8;
            const int col = (wn << 5) + (tn << 4) + lm;
            Ob[(size_t)row * H + col] = f2b(o[tn][j] * inv[j]);
        }
}

// ---------------------------------------------------------------------------
extern "C" void kernel_launch(void* const* d_in, const int* in_sizes, int n_in,
                              void* d_out, int out_size, void* d_ws, size_t ws_size,
                              hipStream_t stream)
{
    (void)in_sizes; (void)n_in; (void)out_size; (void)ws_size;

    const float* q  = (const float*)d_in[0];
    const float* k  = (const float*)d_in[1];
    const float* v  = (const float*)d_in[2];
    const float* Wq = (const float*)d_in[3];
    const float* bq = (const float*)d_in[4];
    const float* Wk = (const float*)d_in[5];
    const float* bk = (const float*)d_in[6];
    const float* Wv = (const float*)d_in[7];
    const float* bv = (const float*)d_in[8];
    const float* Wo = (const float*)d_in[9];
    const float* bo = (const float*)d_in[10];

    constexpr int Bsz = 2, S = 2048, H = 1024;
    constexpr size_t MT = (size_t)Bsz * S;          // 4096 rows
    const size_t planeBytes = MT * H * sizeof(unsigned short);

    char* ws = (char*)d_ws;
    __bf16* qp  = (__bf16*)(ws);
    __bf16* kp  = (__bf16*)(ws + planeBytes);
    __bf16* vp  = (__bf16*)(ws + 2 * planeBytes);
    __bf16* ctx = (__bf16*)(ws + 3 * planeBytes);

    dim3 blk(256);
    dim3 gG(H / 128, (unsigned)(MT / 128));          // (8, 32)

    gemm_bias_kernel<true,  false><<<gG, blk, 0, stream>>>(q, Wq, bq, qp, (int)MT, H, H);
    gemm_bias_kernel<true,  false><<<gG, blk, 0, stream>>>(k, Wk, bk, kp, (int)MT, H, H);
    gemm_bias_kernel<true,  false><<<gG, blk, 0, stream>>>(v, Wv, bv, vp, (int)MT, H, H);

    dim3 gA(S / 64, Bsz * 16);                       // (32, 32)
    attention_kernel<<<gA, blk, 0, stream>>>(qp, kp, vp, ctx);

    gemm_bias_kernel<false, true><<<gG, blk, 0, stream>>>(ctx, Wo, bo, d_out, (int)MT, H, H);
}
